// PKEncoder_87308095193557
// MI455X (gfx1250) — matrix operand order repaired and checked by reading, not compiled, not run on hardware
//
#include <hip/hip_runtime.h>
#include <hip/hip_bf16.h>

#define B_SZ   256
#define T_SEQ  2048
#define HID    64
#define IN_DIM 6
#define EPS    1e-5f

typedef _Float16 v2h  __attribute__((ext_vector_type(2)));
typedef _Float16 v8h  __attribute__((ext_vector_type(8)));
typedef _Float16 v16h __attribute__((ext_vector_type(16)));
typedef float    v8f  __attribute__((ext_vector_type(8)));

#define WMMA_F16(a, b, c) \
    __builtin_amdgcn_wmma_f32_16x16x32_f16(false, (a), false, (b), (short)0, (c), false, false)

__device__ __forceinline__ float xor_shfl(float v, int mask) {
    int lane = (int)(threadIdx.x & 31u);
    int idx  = ((lane ^ mask) << 2);
    return __int_as_float(__builtin_amdgcn_ds_bpermute(idx, __float_as_int(v)));
}

__device__ __forceinline__ float fast_sigmoid(float x) {
    // v_exp_f32 + v_rcp_f32, no IEEE division chain
    return __builtin_amdgcn_rcpf(1.f + __expf(-x));
}

__device__ __forceinline__ float fast_tanh(float x) {
#if __has_builtin(__builtin_amdgcn_tanhf)
    return __builtin_amdgcn_tanhf(x);
#else
    return 1.f - 2.f * __builtin_amdgcn_rcpf(1.f + __expf(2.f * x));
#endif
}

__device__ __forceinline__ v16h mk16(v8h lo, v8h hi) {
    return __builtin_shufflevector(lo, hi, 0,1,2,3,4,5,6,7,8,9,10,11,12,13,14,15);
}

// A-fragment (16x32 f16, rows = M, K = kbase..kbase+31) from row-major array, row stride rs.
// ISA 7.12.2: lanes 0-15 hold row (lane), K = {0..7, 16..23}; lanes 16-31 K = {8..15, 24..31}.
__device__ __forceinline__ v16h load_a(const _Float16* p, int rs, int kbase, int lane) {
    int r  = lane & 15;
    int k0 = kbase + ((lane & 16) ? 8 : 0);
    const _Float16* q = p + r * rs + k0;
    v8h lo = *(const v8h*)(q);
    v8h hi = *(const v8h*)(q + 16);
    return mk16(lo, hi);
}

// B-fragment (32x16 f16, cols = N tile at n0, K = kbase..kbase+31) from W stored as W[n][k]
// (row n contiguous in k). Lanes 0-15: col = lane, K = {0..7, 16..23};
// lanes 16-31: K = {8..15, 24..31} (relative to kbase).
__device__ __forceinline__ v16h load_b(const _Float16* w, int n0, int kbase, int lane) {
    int n  = n0 + (lane & 15);
    int k0 = kbase + ((lane & 16) ? 8 : 0);
    const _Float16* q = w + n * 64 + k0;
    v8h lo = *(const v8h*)(q);
    v8h hi = *(const v8h*)(q + 16);
    return mk16(lo, hi);
}

// ---------------- Kernel 0: weight conversion to f16 (w2 transposed) ----------------
__global__ void PK_wconv(const float* __restrict__ wih, const float* __restrict__ whh,
                         const float* __restrict__ w2,
                         _Float16* __restrict__ wihf, _Float16* __restrict__ whhf,
                         _Float16* __restrict__ w2tf) {
    int i = blockIdx.x * 256 + threadIdx.x;
    if (i < 192 * 64) {
        wihf[i] = (_Float16)wih[i];
        whhf[i] = (_Float16)whh[i];
    }
    if (i < 64 * 64) {
        int k = i >> 6, n = i & 63;           // w2 is (HID, D_PK) = w2[k][n]
        w2tf[n * 64 + k] = (_Float16)w2[i];   // store transposed: row n contiguous in k
    }
}

// ---------------- Kernel 1: x = GELU(LN(drug_ce @ w1 + b1)) -> f16 ----------------
__global__ __launch_bounds__(256) void PK_inproj(const float* __restrict__ in,
                                                 const float* __restrict__ w1,
                                                 const float* __restrict__ b1,
                                                 const float* __restrict__ g1,
                                                 const float* __restrict__ bb1,
                                                 _Float16* __restrict__ xf) {
    int row  = blockIdx.x * 8 + (threadIdx.x >> 5);   // one (b,t) row per wave
    int lane = threadIdx.x & 31;
    float a[IN_DIM];
#pragma unroll
    for (int k = 0; k < IN_DIM; ++k) a[k] = in[(size_t)row * IN_DIM + k];
    int c0 = lane * 2;
    float acc0 = b1[c0], acc1 = b1[c0 + 1];
#pragma unroll
    for (int k = 0; k < IN_DIM; ++k) {
        acc0 += a[k] * w1[k * 64 + c0];
        acc1 += a[k] * w1[k * 64 + c0 + 1];
    }
    float s = acc0 + acc1;
    s += xor_shfl(s, 1); s += xor_shfl(s, 2); s += xor_shfl(s, 4);
    s += xor_shfl(s, 8); s += xor_shfl(s, 16);
    float mean = s * (1.f / 64.f);
    float d0 = acc0 - mean, d1 = acc1 - mean;
    float q = d0 * d0 + d1 * d1;
    q += xor_shfl(q, 1); q += xor_shfl(q, 2); q += xor_shfl(q, 4);
    q += xor_shfl(q, 8); q += xor_shfl(q, 16);
    float rstd = rsqrtf(q * (1.f / 64.f) + EPS);
    float y0 = d0 * rstd * g1[c0] + bb1[c0];
    float y1 = d1 * rstd * g1[c0 + 1] + bb1[c0 + 1];
    y0 = 0.5f * y0 * (1.f + erff(y0 * 0.70710678118654752f));
    y1 = 0.5f * y1 * (1.f + erff(y1 * 0.70710678118654752f));
    v2h pk = { (_Float16)y0, (_Float16)y1 };
    *(v2h*)(xf + (size_t)row * 64 + c0) = pk;
}

// ---------------- Kernel 2: persistent GRU + fused out-proj/LN ----------------
__global__ __launch_bounds__(128, 1) void PK_gru(const _Float16* __restrict__ xf,
                                                 const _Float16* __restrict__ wihf,
                                                 const _Float16* __restrict__ whhf,
                                                 const _Float16* __restrict__ w2tf,
                                                 const float* __restrict__ b_ih,
                                                 const float* __restrict__ b_hh,
                                                 const float* __restrict__ b2,
                                                 const float* __restrict__ ln2g,
                                                 const float* __restrict__ ln2b,
                                                 float* __restrict__ out) {
    __shared__ __align__(16) _Float16 hbuf[2][16 * 64];  // ping-pong f16 mirror of h (4 KB)
    __shared__ float red[4][16][2];
    __shared__ float fin[16][2];

    const int tid  = threadIdx.x;
    const int lane = tid & 31;
    const int wv   = tid >> 5;            // 4 waves: wave wv owns hidden cols [16wv,16wv+16)
    const int m0   = blockIdx.x * 16;     // batch-row tile

    for (int i = tid; i < 16 * 64; i += 128) { hbuf[0][i] = (_Float16)0.f; hbuf[1][i] = (_Float16)0.f; }

    const int ch = wv * 16 + (lane & 15);              // hidden column owned by this lane
    const float br   = b_ih[ch]       + b_hh[ch];      // fused r-gate bias
    const float bz   = b_ih[64 + ch]  + b_hh[64 + ch]; // fused z-gate bias
    const float bin_ = b_ih[128 + ch];
    const float bhn  = b_hh[128 + ch];
    const float bo = b2[ch], lg = ln2g[ch], lb = ln2b[ch];

    // ---- hoist all 14 loop-invariant B-fragments into registers (112 VGPRs) ----
    const v16h Bxr0 = load_b(wihf, wv * 16, 0, lane);
    const v16h Bxr1 = load_b(wihf, wv * 16, 32, lane);
    const v16h Bhr0 = load_b(whhf, wv * 16, 0, lane);
    const v16h Bhr1 = load_b(whhf, wv * 16, 32, lane);
    const v16h Bxz0 = load_b(wihf, 64 + wv * 16, 0, lane);
    const v16h Bxz1 = load_b(wihf, 64 + wv * 16, 32, lane);
    const v16h Bhz0 = load_b(whhf, 64 + wv * 16, 0, lane);
    const v16h Bhz1 = load_b(whhf, 64 + wv * 16, 32, lane);
    const v16h Bxn0 = load_b(wihf, 128 + wv * 16, 0, lane);
    const v16h Bxn1 = load_b(wihf, 128 + wv * 16, 32, lane);
    const v16h Bhn0 = load_b(whhf, 128 + wv * 16, 0, lane);
    const v16h Bhn1 = load_b(whhf, 128 + wv * 16, 32, lane);
    const v16h Bo0  = load_b(w2tf, wv * 16, 0, lane);
    const v16h Bo1  = load_b(w2tf, wv * 16, 32, lane);

    __syncthreads();

    v8f hreg = {};                                      // f32 h-state, WMMA C-layout

    const int    arow     = m0 + (lane & 15);
    const size_t xrowbase = (size_t)arow * T_SEQ * 64;
    const int    ak0      = (lane & 16) ? 8 : 0;

    // prefetch x_0 A-fragments
    v16h ax0, ax1;
    {
        const _Float16* q = xf + xrowbase + ak0;
        ax0 = mk16(*(const v8h*)(q),      *(const v8h*)(q + 16));
        ax1 = mk16(*(const v8h*)(q + 32), *(const v8h*)(q + 48));
    }

    int p = 0;
    for (int t = 0; t < T_SEQ; ++t) {
        // prefetch x_{t+1} (independent of recurrence)
        int tn = (t + 1 < T_SEQ) ? t + 1 : t;
        const _Float16* qn = xf + xrowbase + (size_t)tn * 64 + ak0;
        v8h nx0 = *(const v8h*)(qn),      nx1 = *(const v8h*)(qn + 16);
        v8h nx2 = *(const v8h*)(qn + 32), nx3 = *(const v8h*)(qn + 48);

        // h A-fragments from current ping buffer
        v16h ah0 = load_a(hbuf[p], 64, 0, lane);
        v16h ah1 = load_a(hbuf[p], 64, 32, lane);

        v8f gr = {}, gz = {}, gin = {}, ghn = {};
        gr  = WMMA_F16(ax0, Bxr0, gr);
        gr  = WMMA_F16(ax1, Bxr1, gr);
        gr  = WMMA_F16(ah0, Bhr0, gr);
        gr  = WMMA_F16(ah1, Bhr1, gr);
        gz  = WMMA_F16(ax0, Bxz0, gz);
        gz  = WMMA_F16(ax1, Bxz1, gz);
        gz  = WMMA_F16(ah0, Bhz0, gz);
        gz  = WMMA_F16(ah1, Bhz1, gz);
        gin = WMMA_F16(ax0, Bxn0, gin);
        gin = WMMA_F16(ax1, Bxn1, gin);
        ghn = WMMA_F16(ah0, Bhn0, ghn);
        ghn = WMMA_F16(ah1, Bhn1, ghn);

        const int nb = p ^ 1;
        _Float16* hw = hbuf[nb];
#pragma unroll
        for (int j = 0; j < 8; ++j) {
            float r  = fast_sigmoid(gr[j] + br);
            float z  = fast_sigmoid(gz[j] + bz);
            float n  = fast_tanh(gin[j] + bin_ + r * (ghn[j] + bhn));
            float hn = n + z * (hreg[j] - n);           // (1-z)*n + z*h
            hreg[j]  = hn;
            int row  = (lane & 16) ? (j + 8) : j;       // C-layout row
            hw[row * 64 + ch] = (_Float16)hn;
        }
        __syncthreads();   // (A) h_{t+1} visible to all waves

        // fused out-proj on h_{t+1}
        v16h ao0 = load_a(hbuf[nb], 64, 0, lane);
        v16h ao1 = load_a(hbuf[nb], 64, 32, lane);
        v8f oa = {};
        oa = WMMA_F16(ao0, Bo0, oa);
        oa = WMMA_F16(ao1, Bo1, oa);

        // per-row LN stats: reduce over 16 lanes (this wave's 16 cols), then over 4 waves
        float sarr[8], qarr[8];
#pragma unroll
        for (int j = 0; j < 8; ++j) {
            float v = oa[j] + bo;
            oa[j] = v;
            float s = v, q = v * v;
            s += xor_shfl(s, 1); q += xor_shfl(q, 1);
            s += xor_shfl(s, 2); q += xor_shfl(q, 2);
            s += xor_shfl(s, 4); q += xor_shfl(q, 4);
            s += xor_shfl(s, 8); q += xor_shfl(q, 8);
            sarr[j] = s; qarr[j] = q;
        }
        if ((lane & 15) == 0) {
            int rb = (lane & 16) ? 8 : 0;
#pragma unroll
            for (int j = 0; j < 8; ++j) { red[wv][rb + j][0] = sarr[j]; red[wv][rb + j][1] = qarr[j]; }
        }
        __syncthreads();   // (B)
        if (tid < 16) {
            float s = red[0][tid][0] + red[1][tid][0] + red[2][tid][0] + red[3][tid][0];
            float q = red[0][tid][1] + red[1][tid][1] + red[2][tid][1] + red[3][tid][1];
            float m = s * (1.f / 64.f);
            float var = q * (1.f / 64.f) - m * m;
            fin[tid][0] = m;
            fin[tid][1] = rsqrtf(var + EPS);
        }
        __syncthreads();   // (C)
#pragma unroll
        for (int j = 0; j < 8; ++j) {
            int row = (lane & 16) ? (j + 8) : j;
            float y = (oa[j] - fin[row][0]) * fin[row][1] * lg + lb;
            out[((size_t)(m0 + row) * T_SEQ + t) * 64 + ch] = y;
        }

        p = nb;
        ax0 = mk16(nx0, nx1);
        ax1 = mk16(nx2, nx3);
    }
}

extern "C" void kernel_launch(void* const* d_in, const int* in_sizes, int n_in,
                              void* d_out, int out_size, void* d_ws, size_t ws_size,
                              hipStream_t stream) {
    (void)in_sizes; (void)n_in; (void)out_size; (void)ws_size;
    const float* drug = (const float*)d_in[0];
    const float* w1   = (const float*)d_in[1];
    const float* b1   = (const float*)d_in[2];
    const float* ln1g = (const float*)d_in[3];
    const float* ln1b = (const float*)d_in[4];
    const float* wih  = (const float*)d_in[5];
    const float* bih  = (const float*)d_in[6];
    const float* whh  = (const float*)d_in[7];
    const float* bhh  = (const float*)d_in[8];
    const float* w2   = (const float*)d_in[9];
    const float* b2   = (const float*)d_in[10];
    const float* ln2g = (const float*)d_in[11];
    const float* ln2b = (const float*)d_in[12];
    float* out = (float*)d_out;

    char* ws = (char*)d_ws;
    _Float16* xf   = (_Float16*)ws;                                  // B*T*64 f16 = 67 MB
    _Float16* wihf = (_Float16*)(ws + (size_t)B_SZ * T_SEQ * 64 * 2);
    _Float16* whhf = wihf + 192 * 64;
    _Float16* w2tf = whhf + 192 * 64;

    PK_wconv<<<48, 256, 0, stream>>>(wih, whh, w2, wihf, whhf, w2tf);
    PK_inproj<<<(B_SZ * T_SEQ) / 8, 256, 0, stream>>>(drug, w1, b1, ln1g, ln1b, xf);
    PK_gru<<<B_SZ / 16, 128, 0, stream>>>(xf, wihf, whhf, w2tf, bih, bhh, b2, ln2g, ln2b, out);
}
